// MambaBlock_61856118997525
// MI455X (gfx1250) — matrix operand order repaired and checked by reading, not compiled, not run on hardware
//
#include <hip/hip_runtime.h>
#include <hip/hip_bf16.h>
#include <stdint.h>

// ---- problem dims (fixed by reference) ----
#define NL    4
#define DMOD  256
#define NSTAT 256
#define DI    512
#define DCONV 4
#define DTR   16
#define BATCH 4
#define LSEQ  4096
#define BL    (BATCH * LSEQ)      // 16384 rows
#define XPN   (DTR + 2 * NSTAT)   // 528 = live x_proj output width
#define XPNP  576                 // padded to 36*16 (divisible by 64) for 4-tile blocking

typedef __attribute__((ext_vector_type(16))) _Float16 v16h;
typedef __attribute__((ext_vector_type(8)))  _Float16 v8h;
typedef __attribute__((ext_vector_type(8)))  float    v8f;
typedef __attribute__((ext_vector_type(2)))  float    v2f;
typedef __attribute__((ext_vector_type(2)))  int      v2i;

// pointer types matching the async-LDS builtin signature (global = AS1, LDS = AS3)
typedef __attribute__((address_space(1))) v2i* gptr_v2i;
typedef __attribute__((address_space(3))) v2i* lptr_v2i;

// ---- CDNA5 async LDS support (probe via __has_builtin; fallback = sync path) ----
#if defined(__has_builtin)
#  if __has_builtin(__builtin_amdgcn_global_load_async_to_lds_b64)
#    define HAVE_ASYNC_LDS 1
#  endif
#  if __has_builtin(__builtin_amdgcn_s_wait_asynccnt)
#    define HAVE_WAIT_ASYNC 1
#  endif
#endif

__device__ __forceinline__ float silu_f(float v) { return v / (1.0f + __expf(-v)); }

// ---------------------------------------------------------------------------
// f32 -> f16 conversion (weights once per call + input activations)
// ---------------------------------------------------------------------------
__global__ void k_f32_to_f16(const float* __restrict__ src, _Float16* __restrict__ dst, int n) {
    int i = blockIdx.x * blockDim.x + threadIdx.x;
    if (i < n) dst[i] = (_Float16)src[i];
}

// ---------------------------------------------------------------------------
// f32 -> f16 conversion with row padding: per layer, rows [0,nsrc) copied,
// rows [nsrc,ndst) zero-filled.  dst layout: (NL, ndst, K) f16.
// ---------------------------------------------------------------------------
__global__ void k_f32_to_f16_padrows(const float* __restrict__ src, _Float16* __restrict__ dst,
                                     int nsrc, int ndst, int K, int total) {
    int i = blockIdx.x * blockDim.x + threadIdx.x;       // over NL*ndst*K
    if (i >= total) return;
    const int k     = i % K;
    const int row   = (i / K) % ndst;
    const int layer = i / (K * ndst);
    dst[i] = (row < nsrc) ? (_Float16)src[((size_t)layer * nsrc + row) * K + k] : (_Float16)0.0f;
}

// ---------------------------------------------------------------------------
// WMMA GEMM:  C(M,N) = A(M,K) * W(N,K)^T  (+bias), f16 in / f32 accumulate.
// One wave computes a 16x64 strip (4 N-tiles), reusing the A fragment 4x per
// K-step; K stepped by 32 with v_wmma_f32_16x16x32_f16.
// Fragment K order is contiguous (k = 16*half + s), the same for A and B.
// ---------------------------------------------------------------------------
template <int WPB>
__global__ __launch_bounds__(WPB * 32)
void k_wmma_gemm(const _Float16* __restrict__ A, const _Float16* __restrict__ W,
                 const float* __restrict__ bias,
                 _Float16* __restrict__ outH, float* __restrict__ outF,
                 int M, int N, int K)
{
    const int lane   = threadIdx.x & 31;
    const int wave   = threadIdx.x >> 5;
    const int tilesN = N >> 6;                       // 64-wide strips
    const int tile   = blockIdx.x * WPB + wave;
    const int tm     = (tile / tilesN) << 4;
    const int tn0    = (tile % tilesN) << 6;
    if (tm >= M) return;

    const int ml = lane & 15;        // M row (A) / N col (B,C,D) within tile
    const int kh = lane >> 4;        // K-half select

    const _Float16* arow = A + (size_t)(tm + ml) * K;
    const _Float16* wrow0 = W + (size_t)(tn0 + 0  + ml) * K;
    const _Float16* wrow1 = W + (size_t)(tn0 + 16 + ml) * K;
    const _Float16* wrow2 = W + (size_t)(tn0 + 32 + ml) * K;
    const _Float16* wrow3 = W + (size_t)(tn0 + 48 + ml) * K;

    v8f acc0 = {}, acc1 = {}, acc2 = {}, acc3 = {};
    for (int k0 = 0; k0 < K; k0 += 32) {
        __builtin_prefetch(arow + k0 + 128, 0, 1);   // global_prefetch_b8
        // A 16x32 f16 fragment: lanes<16 hold K 0..15, lanes>=16 hold K 16..31 (matches B)
        v16h af = *(const v16h*)(arow + k0 + kh * 16);
        // B 32x16 f16 fragments: column n = row n of W(N,K); lane-half selects K half
        v16h bf0 = *(const v16h*)(wrow0 + k0 + kh * 16);
        v16h bf1 = *(const v16h*)(wrow1 + k0 + kh * 16);
        v16h bf2 = *(const v16h*)(wrow2 + k0 + kh * 16);
        v16h bf3 = *(const v16h*)(wrow3 + k0 + kh * 16);
        acc0 = __builtin_amdgcn_wmma_f32_16x16x32_f16(false, af, false, bf0, (short)0, acc0, false, false);
        acc1 = __builtin_amdgcn_wmma_f32_16x16x32_f16(false, af, false, bf1, (short)0, acc1, false, false);
        acc2 = __builtin_amdgcn_wmma_f32_16x16x32_f16(false, af, false, bf2, (short)0, acc2, false, false);
        acc3 = __builtin_amdgcn_wmma_f32_16x16x32_f16(false, af, false, bf3, (short)0, acc3, false, false);
    }

#pragma unroll
    for (int s = 0; s < 4; ++s) {
        const v8f acc = (s == 0) ? acc0 : (s == 1) ? acc1 : (s == 2) ? acc2 : acc3;
        const int n   = tn0 + s * 16 + ml;
        const float bv = bias ? bias[n] : 0.0f;
#pragma unroll
        for (int r = 0; r < 8; ++r) {    // VGPR r: M = r + 8*kh
            const int row = tm + r + (kh << 3);
            const float v = acc[r] + bv;
            if (outF) outF[(size_t)row * N + n] = v;
            if (outH) outH[(size_t)row * N + n] = (_Float16)v;
        }
    }
}

// ---------------------------------------------------------------------------
// Depthwise causal conv (DCONV=4) + bias + SiLU, reading u-half of XZ (f32),
// writing f16 U for the x_proj WMMA and the scan.
// ---------------------------------------------------------------------------
__global__ void k_conv_silu(const float* __restrict__ XZ, const float* __restrict__ cw,
                            const float* __restrict__ cb, _Float16* __restrict__ Uh)
{
    int idx = blockIdx.x * blockDim.x + threadIdx.x;     // over BL*DI
    if (idx >= BL * DI) return;
    const int d  = idx % DI;
    const int bl = idx / DI;
    const int l  = bl % LSEQ;
    float acc = cb[d];
#pragma unroll
    for (int j = 0; j < DCONV; ++j) {
        const int t = l - (DCONV - 1) + j;
        if (t >= 0)
            acc += XZ[(size_t)(bl - (DCONV - 1) + j) * (2 * DI) + d] * cw[d * DCONV + j];
    }
    Uh[idx] = (_Float16)silu_f(acc);
}

// ---------------------------------------------------------------------------
// dt = softplus(dt_r @ dt_w^T + dt_b); dt_r = first 16 cols of XDB (stride XPNP).
// ---------------------------------------------------------------------------
__global__ void k_dt_softplus(const float* __restrict__ XDB, const float* __restrict__ dtw,
                              const float* __restrict__ dtb, float* __restrict__ dt)
{
    int idx = blockIdx.x * blockDim.x + threadIdx.x;     // over BL*DI
    if (idx >= BL * DI) return;
    const int d  = idx % DI;
    const int bl = idx / DI;
    const float* r = XDB + (size_t)bl * XPNP;
    float acc = dtb[d];
#pragma unroll
    for (int j = 0; j < DTR; ++j) acc += r[j] * dtw[d * DTR + j];
    dt[idx] = (acc > 20.0f) ? acc : __logf(1.0f + __expf(acc));
}

// ---------------------------------------------------------------------------
// Selective scan. Block = 8 waves, wave = one (b,d) channel, lane holds 8 of
// the 256 states. Per-step B/C row (512 f32 at cols 16..527, shared by block)
// is staged in LDS, double-buffered via global_load_async_to_lds_b64 +
// s_wait_asynccnt. Output fused with D-skip and z-gate, written f16.
// ---------------------------------------------------------------------------
#define SCAN_WAVES 8
__global__ __launch_bounds__(SCAN_WAVES * 32)
void k_selective_scan(const float* __restrict__ XDB, const float* __restrict__ dt,
                      const _Float16* __restrict__ Uh, const float* __restrict__ XZ,
                      const float* __restrict__ A_log, const float* __restrict__ Dskip,
                      _Float16* __restrict__ Yh)
{
    __shared__ float lds_bc[2][2 * NSTAT];               // [buf][B(256)|C(256)]
    const int lane = threadIdx.x & 31;
    const int wave = threadIdx.x >> 5;
    const int pair = blockIdx.x * SCAN_WAVES + wave;     // (b,d) id
    const int b    = pair >> 9;                          // / DI  (uniform per block)
    const int d    = pair & (DI - 1);
    const int t    = threadIdx.x;                        // 256 threads, 2 floats each

    float a[8], h[8];
#pragma unroll
    for (int j = 0; j < 8; ++j) {
        a[j] = -__expf(A_log[d * NSTAT + lane * 8 + j]);
        h[j] = 0.0f;
    }
    const float  dsk  = Dskip[d];
    const size_t base = (size_t)b * LSEQ;

#if HAVE_ASYNC_LDS
    {   // prologue: fill buffer 0 for l=0
        const float* g = XDB + base * XPNP + DTR + 2 * t;
        __builtin_amdgcn_global_load_async_to_lds_b64(
            (gptr_v2i)(uintptr_t)g,
            (lptr_v2i)(uintptr_t)&lds_bc[0][2 * t], 0, 0);
    }
#endif

    for (int l = 0; l < LSEQ; ++l) {
        const int buf = l & 1;
#if HAVE_ASYNC_LDS
        if (l + 1 < LSEQ) {
            const float* g = XDB + (base + l + 1) * XPNP + DTR + 2 * t;
            __builtin_amdgcn_global_load_async_to_lds_b64(
                (gptr_v2i)(uintptr_t)g,
                (lptr_v2i)(uintptr_t)&lds_bc[buf ^ 1][2 * t], 0, 0);
#  if HAVE_WAIT_ASYNC
            __builtin_amdgcn_s_wait_asynccnt(1);         // in-order: buf[l] landed
#  else
            asm volatile("s_wait_asynccnt 1" ::: "memory");
#  endif
        } else {
#  if HAVE_WAIT_ASYNC
            __builtin_amdgcn_s_wait_asynccnt(0);
#  else
            asm volatile("s_wait_asynccnt 0" ::: "memory");
#  endif
        }
        __syncthreads();
#else
        v2f v = *(const v2f*)(XDB + (base + l) * XPNP + DTR + 2 * t);
        __syncthreads();
        lds_bc[buf][2 * t]     = v.x;
        lds_bc[buf][2 * t + 1] = v.y;
        __syncthreads();
#endif
        const size_t row = base + l;
        const float  dtv = dt[row * DI + d];
        const float  uv  = (float)Uh[row * DI + d];
        const float  du  = dtv * uv;
        float acc = 0.0f;
#pragma unroll
        for (int j = 0; j < 8; ++j) {
            const int n  = lane * 8 + j;
            const float dA = __expf(dtv * a[j]);
            h[j] = dA * h[j] + du * lds_bc[buf][n];
            acc += h[j] * lds_bc[buf][NSTAT + n];
        }
#pragma unroll
        for (int off = 16; off >= 1; off >>= 1) acc += __shfl_xor(acc, off, 32);
        if (lane == 0) {
            const float zv = XZ[row * (2 * DI) + DI + d];
            const float y  = (acc + uv * dsk) * silu_f(zv);
            Yh[row * DI + d] = (_Float16)y;
        }
#if HAVE_ASYNC_LDS
        __syncthreads();   // all waves done reading buf before it is refilled
#endif
    }
}

// ---------------------------------------------------------------------------
// host-side orchestration
// ---------------------------------------------------------------------------
extern "C" void kernel_launch(void* const* d_in, const int* in_sizes, int n_in,
                              void* d_out, int out_size, void* d_ws, size_t ws_size,
                              hipStream_t stream)
{
    (void)in_sizes; (void)n_in; (void)out_size; (void)ws_size;
    const float* x     = (const float*)d_in[0];
    const float* ip_w  = (const float*)d_in[1];
    const float* ip_b  = (const float*)d_in[2];
    const float* in_w  = (const float*)d_in[3];
    const float* cw    = (const float*)d_in[4];
    const float* cb    = (const float*)d_in[5];
    const float* xp_w  = (const float*)d_in[6];
    const float* dt_w  = (const float*)d_in[7];
    const float* dt_b  = (const float*)d_in[8];
    const float* A_log = (const float*)d_in[9];
    const float* Dsk   = (const float*)d_in[10];
    const float* out_w = (const float*)d_in[11];
    const float* op_w  = (const float*)d_in[12];
    const float* op_b  = (const float*)d_in[13];
    float* OUT = (float*)d_out;

    // deterministic workspace carve (256B aligned)
    char* p = (char*)d_ws;
    auto carve = [&](size_t bytes) -> char* {
        char* r = p; p += (bytes + 255) & ~(size_t)255; return r;
    };
    _Float16* ipw_h  = (_Float16*)carve((size_t)DMOD * DMOD * 2);
    _Float16* inw_h  = (_Float16*)carve((size_t)NL * 2 * DI * DMOD * 2);
    _Float16* xpw_h  = (_Float16*)carve((size_t)NL * XPNP * DI * 2);   // padded rows
    _Float16* outw_h = (_Float16*)carve((size_t)NL * DMOD * DI * 2);
    _Float16* opw_h  = (_Float16*)carve((size_t)DMOD * DMOD * 2);
    _Float16* Xh     = (_Float16*)carve((size_t)BL * DMOD * 2);
    _Float16* Hh     = (_Float16*)carve((size_t)BL * DMOD * 2);
    float*    XZ     = (float*)   carve((size_t)BL * 2 * DI * 4);
    _Float16* Uh     = (_Float16*)carve((size_t)BL * DI * 2);
    float*    XDB    = (float*)   carve((size_t)BL * XPNP * 4);        // padded stride
    float*    dtf    = (float*)   carve((size_t)BL * DI * 4);
    _Float16* Yh     = (_Float16*)carve((size_t)BL * DI * 2);

    auto cvt = [&](const float* s, _Float16* dh, int n) {
        k_f32_to_f16<<<(n + 255) / 256, 256, 0, stream>>>(s, dh, n);
    };
    cvt(ip_w,  ipw_h,  DMOD * DMOD);
    cvt(in_w,  inw_h,  NL * 2 * DI * DMOD);
    cvt(out_w, outw_h, NL * DMOD * DI);
    cvt(op_w,  opw_h,  DMOD * DMOD);
    cvt(x,     Xh,     BL * DMOD);
    {   // x_proj weights: pad 528 -> 576 rows per layer with zeros
        const int total = NL * XPNP * DI;
        k_f32_to_f16_padrows<<<(total + 255) / 256, 256, 0, stream>>>(
            xp_w, xpw_h, XPN, XPNP, DI, total);
    }

    auto gemm = [&](const _Float16* A, const _Float16* W, const float* bias,
                    _Float16* oh, float* of, int M, int N, int K) {
        const int tiles = (M / 16) * (N / 64);           // 16x64 strip per wave
        k_wmma_gemm<4><<<tiles / 4, 128, 0, stream>>>(A, W, bias, oh, of, M, N, K);
    };

    // input projection -> H (f16)
    gemm(Xh, ipw_h, ip_b, Hh, nullptr, BL, DMOD, DMOD);

    for (int i = 0; i < NL; ++i) {
        gemm(Hh, inw_h + (size_t)i * 2 * DI * DMOD, nullptr, nullptr, XZ, BL, 2 * DI, DMOD);
        k_conv_silu<<<(BL * DI + 255) / 256, 256, 0, stream>>>(
            XZ, cw + (size_t)i * DI * DCONV, cb + (size_t)i * DI, Uh);
        gemm(Uh, xpw_h + (size_t)i * XPNP * DI, nullptr, nullptr, XDB, BL, XPNP, DI);
        k_dt_softplus<<<(BL * DI + 255) / 256, 256, 0, stream>>>(
            XDB, dt_w + (size_t)i * DI * DTR, dt_b + (size_t)i * DI, dtf);
        k_selective_scan<<<(BATCH * DI) / SCAN_WAVES, SCAN_WAVES * 32, 0, stream>>>(
            XDB, dtf, Uh, XZ, A_log + (size_t)i * DI * NSTAT, Dsk + (size_t)i * DI, Yh);
        gemm(Yh, outw_h + (size_t)i * DMOD * DI, nullptr, Hh, nullptr, BL, DMOD, DI);
    }

    // output projection -> d_out (f32, with bias)
    gemm(Hh, opw_h, op_b, nullptr, OUT, BL, DMOD, DMOD);
}
